// GNNWithEdgeFeatures_11991548690480
// MI455X (gfx1250) — matrix operand order repaired and checked by reading, not compiled, not run on hardware
//
#include <hip/hip_runtime.h>

// ---------------------------------------------------------------------------
// GNN with edge features on gfx1250 (MI455X).
//  - bf16 WMMA (v_wmma_f32_16x16x32_bf16) for both dense transforms,
//    with W pre-packed once into the exact CDNA5 B-operand lane layout
//  - wave-per-edge global_atomic_add_f32 scatter (L2-resident accumulators)
//  - global_prefetch_b8 on the streaming edge-feature pass
// ---------------------------------------------------------------------------

typedef __attribute__((ext_vector_type(16))) __bf16 v16bf;
typedef __attribute__((ext_vector_type(8)))  float  v8f;

#define F_NODE 128
#define F_EDGE 32
#define K_TOT  160   // F_NODE + F_EDGE
#define KSTEPS (K_TOT / 32)

// ---------------- utility: zero a float region (graph-capture safe) --------
__global__ void zero_f32(float* __restrict__ p, long long n) {
  long long i = (long long)blockIdx.x * blockDim.x + threadIdx.x;
  long long stride = (long long)gridDim.x * blockDim.x;
  for (; i < n; i += stride) p[i] = 0.0f;
}

// ---------------- edge pass 1: agg[row] += edge_attr ; deg[col] += 1 -------
// one wave (32 lanes) per edge; lane == edge feature index (F_EDGE == 32)
__global__ void edge_pass1(const float* __restrict__ ea,
                           const int* __restrict__ row,
                           const int* __restrict__ col,
                           float* __restrict__ agg,
                           float* __restrict__ deg,
                           int E) {
  long long gid = (long long)blockIdx.x * blockDim.x + threadIdx.x;
  int e    = (int)(gid >> 5);
  int lane = (int)(gid & 31);
  if (e >= E) return;
  int r = row[e];                       // wave-uniform -> scalar load
  if (lane == 0) atomicAdd(&deg[col[e]], 1.0f);
  if (e + 512 < E)                      // stream-ahead hint -> global_prefetch_b8
    __builtin_prefetch(ea + (size_t)(e + 512) * F_EDGE + lane, 0, 1);
  atomicAdd(&agg[(size_t)r * F_EDGE + lane], ea[(size_t)e * F_EDGE + lane]);
}

// ---------------- dinv = rsqrt(deg + 1)  (self-loop => deg+1 > 0 always) ---
__global__ void dinv_kernel(const float* __restrict__ deg,
                            float* __restrict__ dinv, int n) {
  int i = blockIdx.x * blockDim.x + threadIdx.x;
  if (i < n) dinv[i] = rsqrtf(deg[i] + 1.0f);
}

// ---------------- pre-pack W into WMMA B-operand lane layout (bf16) --------
// B 32x16 bf16 layout: lanes 0-15 hold K=[0..7],[16..23], lanes 16-31 hold K=[8..15],[24..31],
// column = lane & 15; dword j of a lane's v16bf = { K=kb+2j+(j>=4?8:0), +1 }.
// Output: Bp[kstep][ntile][lane] : one 32-byte v16bf record per lane.
template <int OUTC>
__global__ void __launch_bounds__(32)
pack_w_bf16(const float* __restrict__ W, unsigned int* __restrict__ Bp) {
  const int lane = threadIdx.x;                 // 0..31
  const int col  = lane & 15;
  const int hi   = (lane >> 4) & 1;
  const int kbB  = blockIdx.x * 32 + (hi ? 8 : 0);
  const int colbase = blockIdx.y * 16;
  unsigned int* dst =
      Bp + ((size_t)(blockIdx.x * (OUTC / 16) + blockIdx.y) * 32 + lane) * 8;
#pragma unroll
  for (int j = 0; j < 8; ++j) {
    __bf16 lo = (__bf16)W[(kbB + 2 * j + (j >= 4 ? 8 : 0))     * OUTC + colbase + col];
    __bf16 hb = (__bf16)W[(kbB + 2 * j + (j >= 4 ? 8 : 0) + 1) * OUTC + colbase + col];
    unsigned int u = (unsigned int)__builtin_bit_cast(unsigned short, lo) |
                     ((unsigned int)__builtin_bit_cast(unsigned short, hb) << 16);
    dst[j] = u;
  }
}

// ---------------- 8 contiguous K-values of A = [A0 | A1] row, optional relu-
template <bool RELU>
__device__ __forceinline__ void load8(const float* __restrict__ A0,
                                      const float* __restrict__ A1,
                                      int row, int k, float* o) {
  const float* p = (k < F_NODE) ? (A0 + (size_t)row * F_NODE + k)
                                : (A1 + (size_t)row * F_EDGE + (k - F_NODE));
  float4 v0 = *(const float4*)(p);
  float4 v1 = *(const float4*)(p + 4);
  o[0] = v0.x; o[1] = v0.y; o[2] = v0.z; o[3] = v0.w;
  o[4] = v1.x; o[5] = v1.y; o[6] = v1.z; o[7] = v1.w;
  if (RELU && k < F_NODE) {            // k is constant after unroll -> folds
#pragma unroll
    for (int i = 0; i < 8; ++i) o[i] = (o[i] > 0.0f) ? o[i] : 0.0f;
  }
}

// ---------------- WMMA GEMM: D[16x16 tile] = concat(A0,A1)[16xK] @ W[KxOUTC]
// one wave per tile; K = 160 = 5 steps of 32 (bf16 in, f32 accumulate);
// B comes pre-packed in register layout (one 32B vector load per step).
template <int OUTC, bool RELU>
__global__ void __launch_bounds__(32)
gemm_wmma_bf16(const float* __restrict__ A0,    // [n, 128]
               const float* __restrict__ A1,    // [n, 32]
               const v16bf* __restrict__ Bp,    // packed W, WMMA lane layout
               float* __restrict__ D,           // [n, OUTC]
               int n) {
  const int lane = threadIdx.x;            // 0..31 (wave32)
  const int m    = lane & 15;              // A row within tile
  const int hi   = (lane >> 4) & 1;        // half-wave select
  const int col  = lane & 15;              // B/D column within tile
  int arow = blockIdx.x * 16 + m;
  if (arow >= n) arow = n - 1;             // clamp reads; stores predicated below

  const v16bf* bbase = Bp + blockIdx.y * 32 + lane;
  v8f c = {};
#pragma unroll
  for (int ks = 0; ks < KSTEPS; ++ks) {
    const int k0 = ks * 32;
    // ---- A 16x32 bf16 layout: lanes 0-15 hold K=[0..7],[16..23];
    //      lanes 16-31 hold K=[8..15],[24..31]  (M = lane & 15)
    float a0[8], a1[8];
    const int kbA = k0 + (hi ? 8 : 0);
    load8<RELU>(A0, A1, arow, kbA,      a0);
    load8<RELU>(A0, A1, arow, kbA + 16, a1);
    v16bf a;
#pragma unroll
    for (int j = 0; j < 8; ++j) {
      a[j]     = (__bf16)a0[j];
      a[8 + j] = (__bf16)a1[j];
    }
    v16bf b = bbase[(size_t)ks * (OUTC / 16) * 32];   // 2x global_load_b128
    c = __builtin_amdgcn_wmma_f32_16x16x32_bf16(
        /*neg_a=*/false, a, /*neg_b=*/false, b,
        /*c_mod=*/(short)0, c, /*reuse_a=*/false, /*reuse_b=*/false);
  }

  // ---- D 16x16 f32 layout: lanes 0-15 rows 0..7, lanes 16-31 rows 8..15
  const int colbase = blockIdx.y * 16;
  const int mbase   = blockIdx.x * 16 + (hi ? 8 : 0);
#pragma unroll
  for (int r = 0; r < 8; ++r) {
    int mm = mbase + r;
    if (mm < n) D[(size_t)mm * OUTC + colbase + col] = c[r];
  }
}

// ---------------- self-loop init: out[i,:] = bias + hpre[i,:] * dinv[i]^2 --
template <int C>
__global__ void self_init(float* __restrict__ out,
                          const float* __restrict__ h,
                          const float* __restrict__ dinv,
                          const float* __restrict__ bias,
                          long long total) {
  long long gid = (long long)blockIdx.x * blockDim.x + threadIdx.x;
  if (gid >= total) return;
  int i = (int)(gid / C);                 // C is power of two -> shift/mask
  int j = (int)(gid % C);
  float di = dinv[i];
  out[gid] = bias[j] + h[gid] * di * di;
}

// ---------------- edge scatter: out[col] += hpre[row] * dinv[row]*dinv[col] -
// one wave per edge; each lane handles VEC contiguous features (C = 32*VEC)
template <int VEC>
__global__ void edge_scatter(float* __restrict__ out,
                             const float* __restrict__ h,
                             const float* __restrict__ dinv,
                             const int* __restrict__ row,
                             const int* __restrict__ col,
                             int E) {
  const int C = 32 * VEC;
  long long gid = (long long)blockIdx.x * blockDim.x + threadIdx.x;
  int e    = (int)(gid >> 5);
  int lane = (int)(gid & 31);
  if (e >= E) return;
  int r = row[e];                          // wave-uniform -> scalar loads
  int d = col[e];
  float nrm = dinv[r] * dinv[d];
  const float* hp = h   + (size_t)r * C + lane * VEC;
  float*       op = out + (size_t)d * C + lane * VEC;
  float v[VEC];
  if constexpr (VEC == 4) {
    float4 t = *(const float4*)hp;
    v[0] = t.x; v[1] = t.y; v[2] = t.z; v[3] = t.w;
  } else {
    float2 t = *(const float2*)hp;
    v[0] = t.x; v[1] = t.y;
  }
#pragma unroll
  for (int i = 0; i < VEC; ++i) atomicAdd(op + i, v[i] * nrm);
}

// ---------------------------------------------------------------------------
extern "C" void kernel_launch(void* const* d_in, const int* in_sizes, int n_in,
                              void* d_out, int out_size, void* d_ws, size_t ws_size,
                              hipStream_t stream) {
  const float* x  = (const float*)d_in[0];   // [N, 128]
  const int*   ei = (const int*)d_in[1];     // [2, E]
  const float* ea = (const float*)d_in[2];   // [E, 32]
  const float* w1 = (const float*)d_in[3];   // [160, 128]
  const float* b1 = (const float*)d_in[4];   // [128]
  const float* w2 = (const float*)d_in[5];   // [160, 64]
  const float* b2 = (const float*)d_in[6];   // [64]

  const int N = in_sizes[0] / F_NODE;
  const int E = in_sizes[1] / 2;
  const int* row = ei;                       // edge_index[0]
  const int* col = ei + E;                   // edge_index[1]

  // workspace layout (dwords); everything is zeroed or fully rewritten per call
  unsigned int* wsu = (unsigned int*)d_ws;
  unsigned int* bp1 = wsu;                                 // 5*8*32*8  = 10240
  unsigned int* bp2 = bp1 + (size_t)KSTEPS * 8 * 32 * 8;   // 5*4*32*8  =  5120
  float* agg  = (float*)(bp2 + (size_t)KSTEPS * 4 * 32 * 8); // [N, 32]
  float* deg  = agg  + (size_t)N * F_EDGE;                 // [N]
  float* dinv = deg  + N;                                  // [N]
  float* h1   = dinv + N;                                  // [N, 128]
  float* o1   = h1   + (size_t)N * 128;                    // [N, 128]
  float* h2   = o1   + (size_t)N * 128;                    // [N, 64]
  float* out  = (float*)d_out;                             // [N, 64]

  const int ethreads = E * 32;                             // 51.2M, fits int
  const int eblocks  = (ethreads + 255) / 256;

  // 0) pack W1/W2 into WMMA B lane layout (tiny, once per call)
  pack_w_bf16<128><<<dim3(KSTEPS, 128 / 16), 32, 0, stream>>>(w1, bp1);
  pack_w_bf16<64 ><<<dim3(KSTEPS, 64 / 16),  32, 0, stream>>>(w2, bp2);

  // 1) zero agg + deg (contiguous N*33 floats)
  zero_f32<<<2048, 256, 0, stream>>>(agg, (long long)N * (F_EDGE + 1));

  // 2) edge feature aggregation + in-degree count
  edge_pass1<<<eblocks, 256, 0, stream>>>(ea, row, col, agg, deg, E);

  // 3) dinv = rsqrt(deg + 1)
  dinv_kernel<<<(N + 255) / 256, 256, 0, stream>>>(deg, dinv, N);

  // 4) h1 = concat(x, agg) @ w1   (WMMA bf16)
  gemm_wmma_bf16<128, false><<<dim3((N + 15) / 16, 128 / 16), 32, 0, stream>>>(
      x, agg, (const v16bf*)bp1, h1, N);

  // 5) o1 = b1 + h1 * dinv^2 (self loops)
  self_init<128><<<(int)(((long long)N * 128 + 255) / 256), 256, 0, stream>>>(
      o1, h1, dinv, b1, (long long)N * 128);

  // 6) o1[col] += h1[row] * norm  (edge scatter, C=128)
  edge_scatter<4><<<eblocks, 256, 0, stream>>>(o1, h1, dinv, row, col, E);

  // 7) h2 = concat(relu(o1), agg) @ w2   (WMMA bf16, fused relu on A-load)
  gemm_wmma_bf16<64, true><<<dim3((N + 15) / 16, 64 / 16), 32, 0, stream>>>(
      o1, agg, (const v16bf*)bp2, h2, N);

  // 8) out = b2 + h2 * dinv^2 (self loops; also initializes poisoned d_out)
  self_init<64><<<(int)(((long long)N * 64 + 255) / 256), 256, 0, stream>>>(
      out, h2, dinv, b2, (long long)N * 64);

  // 9) out[col] += h2[row] * norm  (edge scatter, C=64)
  edge_scatter<2><<<eblocks, 256, 0, stream>>>(out, h2, dinv, row, col, E);
}
